// SANFM_DI_48670569398610
// MI455X (gfx1250) — matrix operand order repaired and checked by reading, not compiled, not run on hardware
//
#include <hip/hip_runtime.h>
#include <hip/hip_bf16.h>
#include <math.h>

typedef __attribute__((ext_vector_type(16))) _Float16 v16h;
typedef __attribute__((ext_vector_type(8)))  float    v8f;

#define LDH 72  // padded leading dim (in halves) for 64-wide f16 tiles

__device__ __forceinline__ int lane_id() { return (int)(threadIdx.x & 31); }

// ---- WMMA fragment loaders (wave32; A and B share a contiguous k order) ----------------
// A fragment: 16x32 f16 tile, rows m0..m0+15, k0..k0+31, from row-major src (ld in halves).
// lane L: row = L&15, half = L>>4 selects K 0..15 vs 16..31; element e -> K = half*16+e.
__device__ __forceinline__ v16h frag_a(const _Float16* p, int ld, int m0, int k0) {
  int lane = lane_id();
  int half = lane >> 4;
  int row  = lane & 15;
  const _Float16* base = p + (size_t)(m0 + row) * ld + k0 + half * 16;
  v16h a;
#pragma unroll
  for (int e = 0; e < 16; ++e) {
    a[e] = base[e];
  }
  return a;
}

// B fragment: 32x16 f16 tile (K x N) from N-major storage src[n][k] (ld in halves).
// lane L: col = L&15, half = L>>4 selects K 0..15 vs 16..31; element e -> K = half*16+e.
__device__ __forceinline__ v16h frag_bT(const _Float16* p, int ld, int n0, int k0) {
  int lane = lane_id();
  int half = lane >> 4;
  int col  = lane & 15;
  const _Float16* base = p + (size_t)(n0 + col) * ld + k0 + half * 16;
  v16h b;
#pragma unroll
  for (int e = 0; e < 16; ++e) b[e] = base[e];
  return b;
}

__device__ __forceinline__ v8f wmma16(v16h a, v16h b, v8f c) {
  return __builtin_amdgcn_wmma_f32_16x16x32_f16(false, a, false, b, (short)0, c, false, false);
}

// =============================================================================
// Kernel 1: fm = batch @ fm_w + fm_b  (one row per block, pure bandwidth)
// =============================================================================
__global__ __launch_bounds__(256) void sanfm_fm(const float* __restrict__ batch,
                                                const float* __restrict__ fm_w,
                                                const float* __restrict__ fm_b,
                                                float* __restrict__ fmOut) {
  const int row = blockIdx.x;
  const int tid = threadIdx.x;
  float s = 0.f;
  for (int k = tid; k < 1538; k += 256) s += batch[(size_t)row * 1538 + k] * fm_w[k];
#pragma unroll
  for (int d = 16; d > 0; d >>= 1) s += __shfl_xor(s, d, 32);
  __shared__ float red[8];
  if ((tid & 31) == 0) red[tid >> 5] = s;
  __syncthreads();
  if (tid == 0) {
    float t = 0.f;
#pragma unroll
    for (int i = 0; i < 8; ++i) t += red[i];
    fmOut[row] = t + fm_b[0];
  }
}

// =============================================================================
// Kernel 2: per 64-row block:
//   inter_DA/DM (K=768 WMMA) -> rowsums + CA/CM -> s/sq (K=64 WMMA) -> X ->
//   Qp = X@Q, Kp = X@K, Vt = (X@V)^T   (all f16 outputs for the attention kernel)
// =============================================================================
__global__ __launch_bounds__(256) void sanfm_interx(
    const float* __restrict__ batch,
    const float* __restrict__ W_DA, const float* __restrict__ b_DA,
    const float* __restrict__ W_DM, const float* __restrict__ b_DM,
    const float* __restrict__ hDA, const float* __restrict__ hDM,
    const float* __restrict__ hCA, const float* __restrict__ hCM,
    const float* __restrict__ fvDA, const float* __restrict__ fvDM,
    const float* __restrict__ fvCA, const float* __restrict__ fvCM,
    const float* __restrict__ Qw, const float* __restrict__ Kw, const float* __restrict__ Vw,
    _Float16* __restrict__ Qp, _Float16* __restrict__ Kp, _Float16* __restrict__ Vt) {
  __shared__ _Float16 regionS[9216];           // phase1: A/B staging; phase3: Xs + matT
  __shared__ _Float16 interS[2 * 64 * LDH];    // inter_DA, inter_DM (f16)
  __shared__ _Float16 hT[2 * 64 * LDH];        // hidden_DA^T, hidden_DM^T (f16)
  __shared__ float scal[256];                  // [sumDA 64][sumDM 64][cA 64][cM 64]

  const int tid = threadIdx.x;
  const int w = tid >> 5, lane = tid & 31;
  const int half = lane >> 4, cl = lane & 15;
  const int f = w >> 2, rs = w & 3;            // field (0=DA,1=DM), 16-row stripe
  const int rowBase = blockIdx.x * 64;

  _Float16* As = regionS;                      // [2][64][32] row-major
  _Float16* Bs = regionS + 4096;               // [2][64n][32k] N-major

  v8f acc[4] = {};

  // ---- phase 1: inter = batch[:, fsel] @ W_f, K loop over 768 in 32-chunks ----
  for (int kk = 0; kk < 768; kk += 32) {
    for (int i = tid; i < 4096; i += 256) {
      int ff = i >> 11, rr = (i >> 5) & 63, k = i & 31;
      As[ff * 2048 + rr * 32 + k] =
          (_Float16)batch[(size_t)(rowBase + rr) * 1538 + ff * 768 + kk + k];
    }
    for (int i = tid; i < 4096; i += 256) {
      int ff = i >> 11, k = (i >> 6) & 31, n = i & 63;
      const float* W = ff ? W_DM : W_DA;
      Bs[ff * 2048 + n * 32 + k] = (_Float16)W[(kk + k) * 64 + n];
    }
    if (kk + 32 < 768)
      __builtin_prefetch(batch + (size_t)(rowBase + (tid & 63)) * 1538 + (tid >> 6) * 768 + kk + 32, 0, 3);
    __syncthreads();
    v16h a = frag_a(As + f * 2048, 32, rs * 16, 0);
#pragma unroll
    for (int t = 0; t < 4; ++t)
      acc[t] = wmma16(a, frag_bT(Bs + f * 2048, 32, t * 16, 0), acc[t]);
    __syncthreads();
  }

  // bias, write inter (f16) to LDS
  {
    const float* bias = f ? b_DM : b_DA;
#pragma unroll
    for (int t = 0; t < 4; ++t) {
      int col = t * 16 + cl;
      float bb = bias[col];
#pragma unroll
      for (int e = 0; e < 8; ++e) {
        int row = rs * 16 + half * 8 + e;
        interS[f * (64 * LDH) + row * LDH + col] = (_Float16)(acc[t][e] + bb);
      }
    }
  }
  __syncthreads();

  // hidden^T staging + row sums + CA/CM scalar columns
  for (int i = tid; i < 8192; i += 256) {
    int ff = i >> 12, k = (i >> 6) & 63, n = i & 63;
    const float* H = ff ? hDM : hDA;
    hT[ff * (64 * LDH) + n * LDH + k] = (_Float16)H[k * 64 + n];
  }
  if (tid < 128) {
    int ff = tid >> 6, r = tid & 63;
    float s = 0.f;
    for (int j = 0; j < 64; ++j) s += (float)interS[ff * (64 * LDH) + r * LDH + j];
    scal[ff * 64 + r] = s;
  } else if (tid < 192) {
    int r = tid - 128;
    scal[128 + r] = batch[(size_t)(rowBase + r) * 1538 + 1536];
  } else {
    int r = tid - 192;
    scal[192 + r] = batch[(size_t)(rowBase + r) * 1538 + 1537];
  }
  __syncthreads();

  // ---- phase 2: s = inter@hidden, sq = inter^2@hidden^2 (fragments squared in-reg), X ----
  const int ch = f;  // reuse waves: column half handled by this wave
  v16h aDA0 = frag_a(interS, LDH, rs * 16, 0);
  v16h aDA1 = frag_a(interS, LDH, rs * 16, 32);
  v16h aDM0 = frag_a(interS + 64 * LDH, LDH, rs * 16, 0);
  v16h aDM1 = frag_a(interS + 64 * LDH, LDH, rs * 16, 32);
  v16h qDA0 = aDA0 * aDA0, qDA1 = aDA1 * aDA1;
  v16h qDM0 = aDM0 * aDM0, qDM1 = aDM1 * aDM1;

  _Float16* Xs = regionS;  // [64][LDH], staging region is free now

#pragma unroll
  for (int t = 0; t < 2; ++t) {
    int n0 = ch * 32 + t * 16;
    v16h bA0 = frag_bT(hT, LDH, n0, 0);
    v16h bA1 = frag_bT(hT, LDH, n0, 32);
    v16h bM0 = frag_bT(hT + 64 * LDH, LDH, n0, 0);
    v16h bM1 = frag_bT(hT + 64 * LDH, LDH, n0, 32);
    v8f sA = {}, sqA = {}, sM = {}, sqM = {};
    sA  = wmma16(aDA0, bA0, sA);        sA  = wmma16(aDA1, bA1, sA);
    sqA = wmma16(qDA0, bA0 * bA0, sqA); sqA = wmma16(qDA1, bA1 * bA1, sqA);
    sM  = wmma16(aDM0, bM0, sM);        sM  = wmma16(aDM1, bM1, sM);
    sqM = wmma16(qDM0, bM0 * bM0, sqM); sqM = wmma16(qDM1, bM1 * bM1, sqM);

    int col = n0 + cl;
    float vda = fvDA[col], vdm = fvDM[col], vca = fvCA[col], vcm = fvCM[col];
    float hca = hCA[col], hcm = hCM[col];  // first row of hidden_CA / hidden_CM
#pragma unroll
    for (int e = 0; e < 8; ++e) {
      int row = rs * 16 + half * 8 + e;
      float sAr = scal[row], sMr = scal[64 + row];
      float cA = scal[128 + row], cM = scal[192 + row];
      float a = sA[e], mm = sM[e];
      float X = 0.5f * (a * a - sqA[e]) + 0.5f * (mm * mm - sqM[e])
              + a  * (vdm * sMr + vca * cA + vcm * cM)
              + mm * (vda * sAr + vca * cA + vcm * cM)
              + (cA * hca) * (vda * sAr + vdm * sMr + vcm * cM)
              + (cM * hcm) * (vda * sAr + vdm * sMr + vca * cA);
      Xs[row * LDH + col] = (_Float16)X;
    }
  }
  __syncthreads();

  // ---- phase 3: Qp / Kp / Vp projections; Vp stored transposed (64 x 8192) ----
  v16h ax0 = frag_a(Xs, LDH, rs * 16, 0);
  v16h ax1 = frag_a(Xs, LDH, rs * 16, 32);
  _Float16* matT = regionS + 64 * LDH;  // [64][LDH], does not overlap Xs

#pragma unroll
  for (int mi = 0; mi < 3; ++mi) {
    const float* M = (mi == 0) ? Qw : (mi == 1) ? Kw : Vw;
    for (int i = tid; i < 4096; i += 256) {
      int k = i >> 6, n = i & 63;
      matT[n * LDH + k] = (_Float16)M[k * 64 + n];
    }
    __syncthreads();
#pragma unroll
    for (int t = 0; t < 2; ++t) {
      int n0 = ch * 32 + t * 16;
      v8f z = {};
      z = wmma16(ax0, frag_bT(matT, LDH, n0, 0), z);
      z = wmma16(ax1, frag_bT(matT, LDH, n0, 32), z);
      int col = n0 + cl;
#pragma unroll
      for (int e = 0; e < 8; ++e) {
        int row = rowBase + rs * 16 + half * 8 + e;
        if (mi == 0)      Qp[(size_t)row * 64 + col] = (_Float16)z[e];
        else if (mi == 1) Kp[(size_t)row * 64 + col] = (_Float16)z[e];
        else              Vt[(size_t)col * 8192 + row] = (_Float16)z[e];
      }
    }
    __syncthreads();
  }
}

// =============================================================================
// Kernel 3: flash attention (128 queries / block, 8 waves x 16-row stripes)
//           + fused MLP tail: h = relu((vw@W1+b1)/sqrt(1+eps)); sigmoid(fm+h@W2+b2)
// =============================================================================
__global__ __launch_bounds__(256) void sanfm_attn(
    const _Float16* __restrict__ Qp, const _Float16* __restrict__ Kp,
    const _Float16* __restrict__ Vt,
    const float* __restrict__ W1, const float* __restrict__ b1,
    const float* __restrict__ W2, const float* __restrict__ b2,
    const float* __restrict__ fm, float* __restrict__ out) {
  __shared__ _Float16 W1t[64 * LDH];          // W1^T as f16
  __shared__ _Float16 Pbuf[8 * 16 * LDH];     // wave-private P / vw staging

  const int tid = threadIdx.x;
  const int w = tid >> 5, lane = tid & 31;
  const int half = lane >> 4, cl = lane & 15;

  for (int i = tid; i < 4096; i += 256) {
    int k = i >> 6, n = i & 63;
    W1t[n * LDH + k] = (_Float16)W1[k * 64 + n];
  }
  __syncthreads();

  const int qRow = blockIdx.x * 128 + w * 16;
  v16h aq0 = frag_a(Qp + (size_t)qRow * 64, 64, 0, 0);
  v16h aq1 = frag_a(Qp + (size_t)qRow * 64, 64, 0, 32);

  float mrow[8], lrow[8];
  v8f O[4] = {};
#pragma unroll
  for (int e = 0; e < 8; ++e) { mrow[e] = -1e30f; lrow[e] = 0.f; }
  _Float16* Pw = Pbuf + w * 16 * LDH;

  for (int kb = 0; kb < 8192; kb += 64) {
    // S = Q @ K^T : Kp row-major is exactly the N-major B layout
    v8f S[4];
#pragma unroll
    for (int t = 0; t < 4; ++t) {
      v8f z = {};
      z = wmma16(aq0, frag_bT(Kp, 64, kb + t * 16, 0), z);
      z = wmma16(aq1, frag_bT(Kp, 64, kb + t * 16, 32), z);
      S[t] = z;
    }
    // online softmax: row max across the 4 tiles then across 16 lanes
    float mx[8];
#pragma unroll
    for (int e = 0; e < 8; ++e)
      mx[e] = fmaxf(fmaxf(S[0][e], S[1][e]), fmaxf(S[2][e], S[3][e]));
#pragma unroll
    for (int d = 1; d < 16; d <<= 1)
#pragma unroll
      for (int e = 0; e < 8; ++e) mx[e] = fmaxf(mx[e], __shfl_xor(mx[e], d, 32));

    float psum[8];
#pragma unroll
    for (int e = 0; e < 8; ++e) {
      float mn = fmaxf(mrow[e], mx[e]);
      float sc = __expf(mrow[e] - mn);
      mrow[e] = mn;
      lrow[e] *= sc;
      psum[e] = 0.f;
#pragma unroll
      for (int t = 0; t < 4; ++t) {
        float p = __expf(S[t][e] - mn);
        psum[e] += p;
        Pw[(half * 8 + e) * LDH + t * 16 + cl] = (_Float16)p;
        O[t][e] *= sc;
      }
    }
#pragma unroll
    for (int d = 1; d < 16; d <<= 1)
#pragma unroll
      for (int e = 0; e < 8; ++e) psum[e] += __shfl_xor(psum[e], d, 32);
#pragma unroll
    for (int e = 0; e < 8; ++e) lrow[e] += psum[e];

    asm volatile("s_wait_dscnt 0" ::: "memory");
    v16h pa0 = frag_a(Pw, LDH, 0, 0);
    v16h pa1 = frag_a(Pw, LDH, 0, 32);
#pragma unroll
    for (int t = 0; t < 4; ++t) {
      O[t] = wmma16(pa0, frag_bT(Vt + kb, 8192, t * 16, 0), O[t]);
      O[t] = wmma16(pa1, frag_bT(Vt + kb, 8192, t * 16, 32), O[t]);
    }
  }

  // normalize, stage vw (f16) back to wave-private LDS for the W1 GEMM
#pragma unroll
  for (int e = 0; e < 8; ++e) {
    float inv = 1.f / lrow[e];
#pragma unroll
    for (int t = 0; t < 4; ++t)
      Pw[(half * 8 + e) * LDH + t * 16 + cl] = (_Float16)(O[t][e] * inv);
  }
  asm volatile("s_wait_dscnt 0" ::: "memory");
  v16h va0 = frag_a(Pw, LDH, 0, 0);
  v16h va1 = frag_a(Pw, LDH, 0, 32);

  const float bninv = rsqrtf(1.0f + 1e-5f);
  float partial[8];
#pragma unroll
  for (int e = 0; e < 8; ++e) partial[e] = 0.f;
#pragma unroll
  for (int t = 0; t < 4; ++t) {
    v8f h = {};
    h = wmma16(va0, frag_bT(W1t, LDH, t * 16, 0), h);
    h = wmma16(va1, frag_bT(W1t, LDH, t * 16, 32), h);
    int col = t * 16 + cl;
    float bb = b1[col], w2 = W2[col];
#pragma unroll
    for (int e = 0; e < 8; ++e) {
      float hv = fmaxf((h[e] + bb) * bninv, 0.f);
      partial[e] += hv * w2;
    }
  }
#pragma unroll
  for (int d = 1; d < 16; d <<= 1)
#pragma unroll
    for (int e = 0; e < 8; ++e) partial[e] += __shfl_xor(partial[e], d, 32);

  if (cl == 0) {
#pragma unroll
    for (int e = 0; e < 8; ++e) {
      int row = qRow + half * 8 + e;
      float x = fm[row] + partial[e] + b2[0];
      out[row] = 1.f / (1.f + __expf(-x));
    }
  }
}

// =============================================================================
extern "C" void kernel_launch(void* const* d_in, const int* in_sizes, int n_in,
                              void* d_out, int out_size, void* d_ws, size_t ws_size,
                              hipStream_t stream) {
  const float* batch = (const float*)d_in[0];
  const float* fm_w  = (const float*)d_in[1];
  const float* fm_b  = (const float*)d_in[2];
  const float* W_DA  = (const float*)d_in[3];
  const float* b_DA  = (const float*)d_in[4];
  const float* W_DM  = (const float*)d_in[5];
  const float* b_DM  = (const float*)d_in[6];
  const float* hDA   = (const float*)d_in[7];
  const float* hDM   = (const float*)d_in[8];
  const float* hCA   = (const float*)d_in[9];
  const float* hCM   = (const float*)d_in[10];
  const float* fvDA  = (const float*)d_in[11];
  const float* fvDM  = (const float*)d_in[12];
  const float* fvCA  = (const float*)d_in[13];
  const float* fvCM  = (const float*)d_in[14];
  const float* Qw    = (const float*)d_in[15];
  const float* Kw    = (const float*)d_in[16];
  const float* Vw    = (const float*)d_in[17];
  const float* W1    = (const float*)d_in[18];
  const float* b1    = (const float*)d_in[19];
  const float* W2    = (const float*)d_in[20];
  const float* b2    = (const float*)d_in[21];

  _Float16* Qp = (_Float16*)d_ws;            // 8192*64 f16
  _Float16* Kp = Qp + 8192 * 64;             // 8192*64 f16
  _Float16* Vt = Kp + 8192 * 64;             // 64*8192 f16 (transposed Vp)
  float* fmBuf = (float*)(Vt + 8192 * 64);   // 8192 f32

  sanfm_fm<<<8192, 256, 0, stream>>>(batch, fm_w, fm_b, fmBuf);
  sanfm_interx<<<128, 256, 0, stream>>>(batch, W_DA, b_DA, W_DM, b_DM, hDA, hDM, hCA, hCM,
                                        fvDA, fvDM, fvCA, fvCM, Qw, Kw, Vw, Qp, Kp, Vt);
  sanfm_attn<<<64, 256, 0, stream>>>(Qp, Kp, Vt, W1, b1, W2, b2, fmBuf, (float*)d_out);
}
